// ElmanRNN_55834574848348
// MI455X (gfx1250) — matrix operand order repaired and checked by reading, not compiled, not run on hardware
//
#include <hip/hip_runtime.h>

typedef __attribute__((ext_vector_type(16))) __bf16 v16bf;
typedef __attribute__((ext_vector_type(8)))  __bf16 v8bf;
typedef __attribute__((ext_vector_type(8)))  float  v8f;

// ---------------------------------------------------------------------------
// Load a 16x32 bf16 A-fragment (row-major source). ISA layout: lane L holds
// row M=L%16; lane<16 -> K = {0..7, 16..23}, lane>=16 -> K = {8..15, 24..31}.
// Caller pre-offsets pointer by kSel. Lowers to two global_load_b128.
// ---------------------------------------------------------------------------
__device__ __forceinline__ v16bf load_a_frag(const __bf16* p) {
  v8bf lo = *(const v8bf*)(p);        // K = kSel + 0..7
  v8bf hi = *(const v8bf*)(p + 16);   // K = kSel + 16..23
  v16bf a;
#pragma unroll
  for (int j = 0; j < 8; ++j) { a[j] = lo[j]; a[j + 8] = hi[j]; }
  return a;
}

// ---------------------------------------------------------------------------
// Pack row-major f32 KxN matrix into WMMA B-fragment-ordered bf16:
// entry((kTile,nTile,lane)) = 16 contiguous bf16 = lane's slice of the
// 32x16 B tile.  col = nTile*16 + lane%16 ; K = kTile*32 + (lane&16?8:0) + {0..7, 16..23}.
// ---------------------------------------------------------------------------
__global__ __launch_bounds__(256) void pack_b_kernel(const float* __restrict__ src,
                                                     __bf16* __restrict__ dst,
                                                     int K, int N) {
  int gid = blockIdx.x * blockDim.x + threadIdx.x;
  int nTiles = N >> 4;
  int total  = (K >> 5) * nTiles * 32;
  if (gid >= total) return;
  int lane  = gid & 31;
  int tile  = gid >> 5;
  int nTile = tile % nTiles;
  int kTile = tile / nTiles;
  int col   = (nTile << 4) + (lane & 15);
  int kBase = (kTile << 5) + ((lane & 16) ? 8 : 0);
  v16bf v;
#pragma unroll
  for (int j = 0; j < 16; ++j)
    v[j] = (__bf16)src[(size_t)(kBase + (j < 8 ? j : j + 8)) * N + col];
  *((v16bf*)dst + gid) = v;
}

// f32 x[B][T][D] -> bf16 xbf[T*B][D]  (row r = t*B + b); coalesced both sides.
__global__ __launch_bounds__(256) void convert_x_kernel(const float* __restrict__ x,
                                                        __bf16* __restrict__ xbf,
                                                        int B, int T, int D) {
  size_t i = (size_t)blockIdx.x * blockDim.x + threadIdx.x;
  size_t total = (size_t)B * T * D;
  if (i >= total) return;
  int d = (int)(i % (size_t)D);
  size_t r = i / (size_t)D;
  int b = (int)(r % (size_t)B);
  int t = (int)(r / (size_t)B);
  xbf[i] = (__bf16)x[((size_t)b * T + t) * D + d];
}

__global__ __launch_bounds__(256) void zero_bf16_kernel(__bf16* p, int nDwords) {
  int i = blockIdx.x * blockDim.x + threadIdx.x;
  if (i < nDwords) ((unsigned int*)p)[i] = 0u;
}

// ---------------------------------------------------------------------------
// xw = xbf @ W + b   (M=T*B, K=D, N=H), f32 out.
// 8 waves/block; wave computes a 64x16 strip (4 stacked 16x16 tiles reusing
// one B fragment).  2-stage software pipeline; B walks a constant-stride ptr.
// ---------------------------------------------------------------------------
__global__ __launch_bounds__(256) void xw_gemm_kernel(const __bf16* __restrict__ xbf,
                                                      const __bf16* __restrict__ WP,
                                                      const float*  __restrict__ bias,
                                                      float* __restrict__ xw,
                                                      int D, int H) {
  const int lane  = threadIdx.x & 31;
  const int wave  = threadIdx.x >> 5;
  const int mBase = blockIdx.x * 64;
  const int nBase = blockIdx.y * 128 + wave * 16;
  const int nTile = nBase >> 4;
  const int nTiles = H >> 4;

  v8f acc0 = {}, acc1 = {}, acc2 = {}, acc3 = {};
  const int kSel = (lane & 16) ? 8 : 0;
  const __bf16* aRow = xbf + (size_t)(mBase + (lane & 15)) * D + kSel;
  // packed-B: one 32B entry per (kTile, nTile, lane); kTile stride is constant
  const size_t  bStride = (size_t)nTiles * 32 * 16;           // elements per kTile
  const __bf16* bp = WP + ((size_t)nTile * 32 + lane) * 16;   // kTile = 0

  // -- pipeline prologue: fragments for k = 0
  v16bf bf = *(const v16bf*)bp;
  v16bf a0 = load_a_frag(aRow);
  v16bf a1 = load_a_frag(aRow + (size_t)16 * D);
  v16bf a2 = load_a_frag(aRow + (size_t)32 * D);
  v16bf a3 = load_a_frag(aRow + (size_t)48 * D);

  for (int k = 0; k < D - 32; k += 32) {
    // issue next-iteration loads before consuming current fragments
    bp += bStride;
    v16bf bfn = *(const v16bf*)bp;
    v16bf a0n = load_a_frag(aRow + k + 32);
    v16bf a1n = load_a_frag(aRow + (size_t)16 * D + k + 32);
    v16bf a2n = load_a_frag(aRow + (size_t)32 * D + k + 32);
    v16bf a3n = load_a_frag(aRow + (size_t)48 * D + k + 32);
    acc0 = __builtin_amdgcn_wmma_f32_16x16x32_bf16(false, a0, false, bf, (short)0, acc0, false, false);
    acc1 = __builtin_amdgcn_wmma_f32_16x16x32_bf16(false, a1, false, bf, (short)0, acc1, false, false);
    acc2 = __builtin_amdgcn_wmma_f32_16x16x32_bf16(false, a2, false, bf, (short)0, acc2, false, false);
    acc3 = __builtin_amdgcn_wmma_f32_16x16x32_bf16(false, a3, false, bf, (short)0, acc3, false, false);
    bf = bfn; a0 = a0n; a1 = a1n; a2 = a2n; a3 = a3n;
  }
  // -- pipeline epilogue
  acc0 = __builtin_amdgcn_wmma_f32_16x16x32_bf16(false, a0, false, bf, (short)0, acc0, false, false);
  acc1 = __builtin_amdgcn_wmma_f32_16x16x32_bf16(false, a1, false, bf, (short)0, acc1, false, false);
  acc2 = __builtin_amdgcn_wmma_f32_16x16x32_bf16(false, a2, false, bf, (short)0, acc2, false, false);
  acc3 = __builtin_amdgcn_wmma_f32_16x16x32_bf16(false, a3, false, bf, (short)0, acc3, false, false);

  const int col  = nBase + (lane & 15);
  const float bv = bias[col];
  const int rOff = (lane & 16) ? 8 : 0;   // C/D layout: VGPR v -> M = v (+8 for hi lanes)
  v8f accs[4] = {acc0, acc1, acc2, acc3};
#pragma unroll
  for (int mt = 0; mt < 4; ++mt) {
    float* o = xw + (size_t)(mBase + mt * 16 + rOff) * H + col;
#pragma unroll
    for (int v = 0; v < 8; ++v) o[(size_t)v * H] = accs[mt][v] + bv;
  }
}

// ---------------------------------------------------------------------------
// One scan step:  hNext = tanh(xw_t + hPrev @ Wh).  M fixed = 64 (= B).
// 4 waves/block, wave = 64x16 strip, grid = H/64.  Same pipeline structure.
// ---------------------------------------------------------------------------
__global__ __launch_bounds__(128) void rnn_step_kernel(const __bf16* __restrict__ hPrev,
                                                       const float*  __restrict__ xw_t,
                                                       const __bf16* __restrict__ WhP,
                                                       __bf16* __restrict__ hNext,
                                                       float*  __restrict__ outF,
                                                       int H, int isFinal) {
  const int lane  = threadIdx.x & 31;
  const int wave  = threadIdx.x >> 5;
  const int nBase = blockIdx.x * 64 + wave * 16;
  const int nTile = nBase >> 4;
  const int nTiles = H >> 4;

  v8f acc0 = {}, acc1 = {}, acc2 = {}, acc3 = {};
  const int kSel = (lane & 16) ? 8 : 0;
  const __bf16* aRow = hPrev + (size_t)(lane & 15) * H + kSel;
  const size_t  bStride = (size_t)nTiles * 32 * 16;
  const __bf16* bp = WhP + ((size_t)nTile * 32 + lane) * 16;

  v16bf bf = *(const v16bf*)bp;
  v16bf a0 = load_a_frag(aRow);
  v16bf a1 = load_a_frag(aRow + (size_t)16 * H);
  v16bf a2 = load_a_frag(aRow + (size_t)32 * H);
  v16bf a3 = load_a_frag(aRow + (size_t)48 * H);

  for (int k = 0; k < H - 32; k += 32) {
    bp += bStride;
    v16bf bfn = *(const v16bf*)bp;
    v16bf a0n = load_a_frag(aRow + k + 32);
    v16bf a1n = load_a_frag(aRow + (size_t)16 * H + k + 32);
    v16bf a2n = load_a_frag(aRow + (size_t)32 * H + k + 32);
    v16bf a3n = load_a_frag(aRow + (size_t)48 * H + k + 32);
    acc0 = __builtin_amdgcn_wmma_f32_16x16x32_bf16(false, a0, false, bf, (short)0, acc0, false, false);
    acc1 = __builtin_amdgcn_wmma_f32_16x16x32_bf16(false, a1, false, bf, (short)0, acc1, false, false);
    acc2 = __builtin_amdgcn_wmma_f32_16x16x32_bf16(false, a2, false, bf, (short)0, acc2, false, false);
    acc3 = __builtin_amdgcn_wmma_f32_16x16x32_bf16(false, a3, false, bf, (short)0, acc3, false, false);
    bf = bfn; a0 = a0n; a1 = a1n; a2 = a2n; a3 = a3n;
  }
  acc0 = __builtin_amdgcn_wmma_f32_16x16x32_bf16(false, a0, false, bf, (short)0, acc0, false, false);
  acc1 = __builtin_amdgcn_wmma_f32_16x16x32_bf16(false, a1, false, bf, (short)0, acc1, false, false);
  acc2 = __builtin_amdgcn_wmma_f32_16x16x32_bf16(false, a2, false, bf, (short)0, acc2, false, false);
  acc3 = __builtin_amdgcn_wmma_f32_16x16x32_bf16(false, a3, false, bf, (short)0, acc3, false, false);

  const int col  = nBase + (lane & 15);
  const int rOff = (lane & 16) ? 8 : 0;
  v8f accs[4] = {acc0, acc1, acc2, acc3};
#pragma unroll
  for (int mt = 0; mt < 4; ++mt) {
#pragma unroll
    for (int v = 0; v < 8; ++v) {
      int m = mt * 16 + rOff + v;
      float val = tanhf(accs[mt][v] + xw_t[(size_t)m * H + col]);
      hNext[(size_t)m * H + col] = (__bf16)val;
      if (isFinal) outF[(size_t)m * H + col] = val;
    }
  }
}

// ---------------------------------------------------------------------------
extern "C" void kernel_launch(void* const* d_in, const int* in_sizes, int n_in,
                              void* d_out, int out_size, void* d_ws, size_t ws_size,
                              hipStream_t stream) {
  const float* x  = (const float*)d_in[0];   // [B][T][D]
  const float* W  = (const float*)d_in[1];   // [D][H]
  const float* Wh = (const float*)d_in[2];   // [H][H]
  const float* b  = (const float*)d_in[3];   // [H]
  const int B = 64, T = 256, D = 2048, H = 2048;
  const int TB = T * B;
  (void)in_sizes; (void)n_in; (void)out_size; (void)ws_size;

  char* ws = (char*)d_ws;
  size_t off = 0;
  __bf16* xbf = (__bf16*)(ws + off); off += (size_t)TB * D * sizeof(__bf16);  //  64 MB
  __bf16* WP  = (__bf16*)(ws + off); off += (size_t)D  * H * sizeof(__bf16);  //   8 MB
  __bf16* WhP = (__bf16*)(ws + off); off += (size_t)H  * H * sizeof(__bf16);  //   8 MB
  float*  xw  = (float*)(ws + off);  off += (size_t)TB * H * sizeof(float);   // 128 MB
  __bf16* h0  = (__bf16*)(ws + off); off += (size_t)B  * H * sizeof(__bf16);
  __bf16* h1  = (__bf16*)(ws + off); off += (size_t)B  * H * sizeof(__bf16);

  // 1) x -> bf16, transposed to [T*B][D]
  {
    size_t total = (size_t)TB * D;
    convert_x_kernel<<<(unsigned)((total + 255) / 256), 256, 0, stream>>>(x, xbf, B, T, D);
  }
  // 2) pack W and Wh into B-fragment order (bf16)
  {
    int total = (D >> 5) * (H >> 4) * 32;
    pack_b_kernel<<<(total + 255) / 256, 256, 0, stream>>>(W, WP, D, H);
  }
  {
    int total = (H >> 5) * (H >> 4) * 32;
    pack_b_kernel<<<(total + 255) / 256, 256, 0, stream>>>(Wh, WhP, H, H);
  }
  // 3) h0 = 0
  {
    int nDwords = (B * H) / 2;
    zero_bf16_kernel<<<(nDwords + 255) / 256, 256, 0, stream>>>(h0, nDwords);
  }
  // 4) xw = x@W + b  (one big WMMA GEMM)
  xw_gemm_kernel<<<dim3(TB / 64, H / 128), 256, 0, stream>>>(xbf, WP, b, xw, D, H);

  // 5) serial scan: 256 dependent step-GEMMs, Wh stays L2-resident
  __bf16* cur = h0;
  __bf16* nxt = h1;
  for (int t = 0; t < T; ++t) {
    rnn_step_kernel<<<H / 64, 128, 0, stream>>>(cur, xw + (size_t)t * B * H, WhP,
                                                nxt, (float*)d_out, H, (t == T - 1) ? 1 : 0);
    __bf16* tmp = cur; cur = nxt; nxt = tmp;
  }
}
